// GraphAttention_15865609191735
// MI455X (gfx1250) — matrix operand order repaired and checked by reading, not compiled, not run on hardware
//
#include <hip/hip_runtime.h>
#include <hip/hip_bf16.h>

typedef __attribute__((ext_vector_type(16))) __bf16       v16bf;
typedef __attribute__((ext_vector_type(8)))  float        v8f;
typedef __attribute__((ext_vector_type(8)))  unsigned int v8u;

constexpr int Bsz = 2, Nsz = 8192, FIN = 128, FOUT = 64;

// round-to-nearest-even f32 -> bf16 bits (cold paths only)
static __device__ __forceinline__ unsigned short f2bfu(float x) {
  unsigned u = __builtin_bit_cast(unsigned, x);
  unsigned r = u + 0x7FFFu + ((u >> 16) & 1u);
  return (unsigned short)(r >> 16);
}
static __device__ __forceinline__ __bf16 f2bf(float x) {
  unsigned short s = f2bfu(x);
  return __builtin_bit_cast(__bf16, s);
}
// truncating f32 -> bf16 bits (hot path; folds into d16_hi stores)
static __device__ __forceinline__ unsigned short f2bfu_rz(float x) {
  return (unsigned short)(__builtin_bit_cast(unsigned, x) >> 16);
}
// load 16 bf16 (8 dwords) as a WMMA operand
static __device__ __forceinline__ v16bf ld_bf16x16(const unsigned int* p) {
  uint4 a = *(const uint4*)p;
  uint4 b = *(const uint4*)(p + 4);
  v8u t; t[0]=a.x; t[1]=a.y; t[2]=a.z; t[3]=a.w; t[4]=b.x; t[5]=b.y; t[6]=b.z; t[7]=b.w;
  return __builtin_bit_cast(v16bf, t);
}

// ---------------------------------------------------------------------------
// Stage 1: q/k/v = 2-layer pointwise MLPs. One wave per row (wave32).
//   q  : f32 [B*N][64]
//   kp : u32 [B][N/16][16][32]    bf16 feature-pairs => WMMA B-operand layout
//   vp : bf16[B][N/32][64][16][2] row-pairs packed   => WMMA B-operand layout
// ---------------------------------------------------------------------------
__global__ __launch_bounds__(256) void qkv_kernel(
    const float* __restrict__ feat,
    const float* __restrict__ W11, const float* __restrict__ b11,
    const float* __restrict__ W12, const float* __restrict__ b12,
    const float* __restrict__ W21, const float* __restrict__ b21,
    const float* __restrict__ W22, const float* __restrict__ b22,
    const float* __restrict__ W31, const float* __restrict__ b31,
    const float* __restrict__ W32, const float* __restrict__ b32,
    float* __restrict__ q, unsigned int* __restrict__ kp,
    unsigned short* __restrict__ vp)
{
  __shared__ float xs[8][FIN];
  __shared__ float hs[8][FOUT];
  __shared__ float os[8][FOUT];
  const int w    = threadIdx.x >> 5;
  const int lane = threadIdx.x & 31;
  const int row  = blockIdx.x * 8 + w;       // 0 .. B*N-1 (grid sized exactly)
  const int b    = row / Nsz, n = row % Nsz;

  for (int t = lane; t < FIN; t += 32) xs[w][t] = feat[(size_t)row * FIN + t];
  __syncthreads();

  for (int m = 0; m < 3; ++m) {
    const float* W1 = m == 0 ? W11 : (m == 1 ? W21 : W31);
    const float* B1 = m == 0 ? b11 : (m == 1 ? b21 : b31);
    const float* W2 = m == 0 ? W12 : (m == 1 ? W22 : W32);
    const float* B2 = m == 0 ? b12 : (m == 1 ? b22 : b32);

    for (int jj = lane; jj < FOUT; jj += 32) {
      float a = B1[jj];
#pragma unroll
      for (int f = 0; f < FIN; ++f) a += xs[w][f] * W1[f * FOUT + jj];
      hs[w][jj] = fmaxf(a, 0.f);
    }
    __syncthreads();
    for (int jj = lane; jj < FOUT; jj += 32) {
      float a = B2[jj];
#pragma unroll
      for (int f = 0; f < FOUT; ++f) a += hs[w][f] * W2[f * FOUT + jj];
      float o = fmaxf(a, 0.f);
      if (m == 0) {
        q[(size_t)row * FOUT + jj] = o;
      } else if (m == 2) {
        size_t vi = ((((size_t)b * (Nsz / 32) + (n >> 5)) * FOUT + jj) * 16 +
                     (((n >> 1) & 3) | (((n >> 1) & 4) << 1) | (((n >> 1) & 8) >> 1))) * 2 + (n & 1);
        vp[vi] = f2bfu(o);
      } else {
        os[w][jj] = o;
      }
    }
    __syncthreads();
    if (m == 1) {  // pack K feature-pairs: one dword per lane (fp = lane)
      unsigned lo = f2bfu(os[w][2 * lane]);
      unsigned hx = f2bfu(os[w][2 * lane + 1]);
      size_t ki = (((size_t)b * (Nsz / 16) + (n >> 4)) * 16 + (n & 15)) * 32 + ((lane & 19) | ((lane & 4) << 1) | ((lane & 8) >> 1));
      kp[ki] = lo | (hx << 16);
    }
    __syncthreads();
  }
}

// ---------------------------------------------------------------------------
// Stage 2: fused flash attention, relu(S * x_weight) softmax, exp2 domain.
// One wave per 16-row tile; 64-column chunks; 18 WMMAs per chunk.
// Row-sum folded into WMMA via a ones-column accumulator (O4).
// ---------------------------------------------------------------------------
__global__ __launch_bounds__(128) void attn_kernel(
    const float* __restrict__ xw,
    const float* __restrict__ q,
    const unsigned int* __restrict__ kp,
    const unsigned int* __restrict__ vp32,
    float* __restrict__ out)
{
  __shared__ unsigned short Pbuf[4][16 * 64];  // per-wave P tile (16x64 bf16)
  const int wv = threadIdx.x >> 5, lane = threadIdx.x & 31;
  const int ml = lane & 15, hi = lane >> 4;
  const int gid = blockIdx.x * 4 + wv;           // 0..1023
  const int bb = gid / (Nsz / 16);
  const int i0 = (gid % (Nsz / 16)) * 16;
  unsigned short* Pw = &Pbuf[wv][0];

  // ---- Q tile as two 16x32 bf16 A operands, pre-scaled by log2(e) so the
  //      whole softmax runs in the exp2 domain (relu commutes with +scale) ----
  constexpr float LOG2E = 1.4426950408889634f;
  const float* qrow = q + (size_t)(bb * Nsz + i0 + ml) * FOUT;
  v16bf qa0, qa1;
#pragma unroll
  for (int e = 0; e < 8; ++e) {
    qa0[e]     = f2bf(qrow[8 * hi + e] * LOG2E);
    qa0[e + 8] = f2bf(qrow[16 + 8 * hi + e] * LOG2E);
    qa1[e]     = f2bf(qrow[32 + 8 * hi + e] * LOG2E);
    qa1[e + 8] = f2bf(qrow[48 + 8 * hi + e] * LOG2E);
  }

  // ---- constant "ones-column" B operand: B[k][n] = (n==0) ----
  v16bf ones;
  {
    __bf16 oe = f2bf(ml == 0 ? 1.f : 0.f);
#pragma unroll
    for (int e = 0; e < 16; ++e) ones[e] = oe;
  }

  const float* xwr[8];
#pragma unroll
  for (int r = 0; r < 8; ++r)
    xwr[r] = xw + ((size_t)bb * Nsz + i0 + r + 8 * hi) * (size_t)Nsz + ml;

  float mrow[8];
  v8f O[4], O4;
#pragma unroll
  for (int r = 0; r < 8; ++r) { mrow[r] = -3.0e38f; O4[r] = 0.f; }
#pragma unroll
  for (int ot = 0; ot < 4; ++ot)
#pragma unroll
    for (int r = 0; r < 8; ++r) O[ot][r] = 0.f;

  const size_t kbase = (size_t)bb * (Nsz / 16);
  const size_t vbase = (size_t)bb * (Nsz / 32);
  v8f z;
#pragma unroll
  for (int r = 0; r < 8; ++r) z[r] = 0.f;

#pragma unroll 1
  for (int j0 = 0; j0 < Nsz; j0 += 64) {
    // ---- S = Q K^T for this 16x64 block (8 WMMAs) ----
    v8f S[4];
#pragma unroll
    for (int ct = 0; ct < 4; ++ct) {
      const unsigned int* kptr =
          kp + ((kbase + (j0 >> 4) + ct) * 16 + ml) * 32 + 8 * hi;
      v16bf bk = ld_bf16x16(kptr);
      S[ct] = __builtin_amdgcn_wmma_f32_16x16x32_bf16(false, qa0, false, bk, (short)0, z, false, false);
      bk = ld_bf16x16(kptr + 16);
      S[ct] = __builtin_amdgcn_wmma_f32_16x16x32_bf16(false, qa1, false, bk, (short)0, S[ct], false, false);
    }
    // ---- x_weight * relu, online softmax in exp2 domain ----
#pragma unroll
    for (int r = 0; r < 8; ++r) {
      float x0 = fmaxf(S[0][r] * __builtin_nontemporal_load(xwr[r] + j0), 0.f);
      float x1 = fmaxf(S[1][r] * __builtin_nontemporal_load(xwr[r] + j0 + 16), 0.f);
      float x2 = fmaxf(S[2][r] * __builtin_nontemporal_load(xwr[r] + j0 + 32), 0.f);
      float x3 = fmaxf(S[3][r] * __builtin_nontemporal_load(xwr[r] + j0 + 48), 0.f);
      float t = fmaxf(fmaxf(x0, x1), fmaxf(x2, x3));
#pragma unroll
      for (int d = 1; d < 16; d <<= 1) t = fmaxf(t, __shfl_xor(t, d, 32));
      float mn    = fmaxf(mrow[r], t);
      float scale = __builtin_amdgcn_exp2f(mrow[r] - mn);
      mrow[r] = mn;
#pragma unroll
      for (int ot = 0; ot < 4; ++ot) O[ot][r] *= scale;
      O4[r] *= scale;
      Pw[(r + 8 * hi) * 64 + ml]      = f2bfu_rz(__builtin_amdgcn_exp2f(x0 - mn));
      Pw[(r + 8 * hi) * 64 + 16 + ml] = f2bfu_rz(__builtin_amdgcn_exp2f(x1 - mn));
      Pw[(r + 8 * hi) * 64 + 32 + ml] = f2bfu_rz(__builtin_amdgcn_exp2f(x2 - mn));
      Pw[(r + 8 * hi) * 64 + 48 + ml] = f2bfu_rz(__builtin_amdgcn_exp2f(x3 - mn));
    }
    asm volatile("s_wait_dscnt 0x0" ::: "memory");  // in-wave LDS store->load
    // ---- re-read P from LDS as two A operands (C->A transpose) ----
    v16bf pa0, pa1;
    {
      uint4 a0 = *(const uint4*)(Pw + ml * 64 + 8 * hi);
      uint4 a1 = *(const uint4*)(Pw + ml * 64 + 16 + 8 * hi);
      uint4 a2 = *(const uint4*)(Pw + ml * 64 + 32 + 8 * hi);
      uint4 a3 = *(const uint4*)(Pw + ml * 64 + 48 + 8 * hi);
      v8u t0; t0[0]=a0.x; t0[1]=a0.y; t0[2]=a0.z; t0[3]=a0.w;
              t0[4]=a1.x; t0[5]=a1.y; t0[6]=a1.z; t0[7]=a1.w;
      v8u t1; t1[0]=a2.x; t1[1]=a2.y; t1[2]=a2.z; t1[3]=a2.w;
              t1[4]=a3.x; t1[5]=a3.y; t1[6]=a3.z; t1[7]=a3.w;
      pa0 = __builtin_bit_cast(v16bf, t0);
      pa1 = __builtin_bit_cast(v16bf, t1);
    }
    // ---- O += P * V (8 WMMAs) and O4 += P * ones (2 WMMAs) ----
    const size_t c0 = vbase + (j0 >> 5);
#pragma unroll
    for (int ot = 0; ot < 4; ++ot) {
      v16bf vb0 = ld_bf16x16(vp32 + ((c0       * 64 + ot * 16 + ml) * 16 + 8 * hi));
      v16bf vb1 = ld_bf16x16(vp32 + (((c0 + 1) * 64 + ot * 16 + ml) * 16 + 8 * hi));
      O[ot] = __builtin_amdgcn_wmma_f32_16x16x32_bf16(false, pa0, false, vb0, (short)0, O[ot], false, false);
      O[ot] = __builtin_amdgcn_wmma_f32_16x16x32_bf16(false, pa1, false, vb1, (short)0, O[ot], false, false);
    }
    O4 = __builtin_amdgcn_wmma_f32_16x16x32_bf16(false, pa0, false, ones, (short)0, O4, false, false);
    O4 = __builtin_amdgcn_wmma_f32_16x16x32_bf16(false, pa1, false, ones, (short)0, O4, false, false);
  }
  // ---- finalize: broadcast row-sum from column 0, 1/sum, relu, store ----
#pragma unroll
  for (int r = 0; r < 8; ++r) {
    float sum = __shfl(O4[r], hi * 16, 32);  // C layout: col 0 lives in lane hi*16
    float inv = 1.0f / sum;
#pragma unroll
    for (int ot = 0; ot < 4; ++ot) {
      float v = fmaxf(O[ot][r] * inv, 0.f);
      out[(size_t)(bb * Nsz + i0 + r + 8 * hi) * FOUT + ot * 16 + ml] = v;
    }
  }
}

extern "C" void kernel_launch(void* const* d_in, const int* in_sizes, int n_in,
                              void* d_out, int out_size, void* d_ws, size_t ws_size,
                              hipStream_t stream) {
  (void)in_sizes; (void)n_in; (void)out_size; (void)ws_size;
  const float* feat = (const float*)d_in[0];
  const float* xw   = (const float*)d_in[1];
  const float* W11  = (const float*)d_in[2];  const float* b11 = (const float*)d_in[3];
  const float* W12  = (const float*)d_in[4];  const float* b12 = (const float*)d_in[5];
  const float* W21  = (const float*)d_in[6];  const float* b21 = (const float*)d_in[7];
  const float* W22  = (const float*)d_in[8];  const float* b22 = (const float*)d_in[9];
  const float* W31  = (const float*)d_in[10]; const float* b31 = (const float*)d_in[11];
  const float* W32  = (const float*)d_in[12]; const float* b32 = (const float*)d_in[13];
  float* out = (float*)d_out;

  char* ws = (char*)d_ws;
  float*          q  = (float*)ws;                                   // 4 MB
  unsigned int*   kp = (unsigned int*)(ws + (size_t)Bsz*Nsz*FOUT*4); // 2 MB
  unsigned short* vp = (unsigned short*)(ws + (size_t)Bsz*Nsz*FOUT*4
                                            + (size_t)Bsz*Nsz*32*4); // 2 MB

  qkv_kernel<<<(Bsz * Nsz) / 8, 256, 0, stream>>>(
      feat, W11, b11, W12, b12, W21, b21, W22, b22, W31, b31, W32, b32,
      q, kp, vp);
  attn_kernel<<<(Bsz * (Nsz / 16)) / 4, 128, 0, stream>>>(
      xw, q, kp, (const unsigned int*)vp, out);
}
